// AdaptiveSparsityGate_62182536512342
// MI455X (gfx1250) — matrix operand order repaired and checked by reading, not compiled, not run on hardware
//
#include <hip/hip_runtime.h>
#include <hip/hip_bf16.h>
#include <math.h>

#define DIM   1024
#define SEQ   32768
#define KMAXC 103

typedef __attribute__((ext_vector_type(16))) __bf16 v16bf;
typedef __attribute__((ext_vector_type(8)))  float  v8f;

__device__ __forceinline__ unsigned keyOf(float f) {
    unsigned u = __float_as_uint(f);
    return (u & 0x80000000u) ? ~u : (u | 0x80000000u);  // order-preserving float->uint
}

// ---------------- Stage 1: column-sum partials of x ----------------
__global__ __launch_bounds__(256) void colsum_kernel(const float* __restrict__ x,
                                                     float* __restrict__ partial) {
    const int c  = blockIdx.x * 256 + threadIdx.x;   // column
    const int r0 = blockIdx.y * 256;                 // row chunk
    float s = 0.0f;
    for (int r = 0; r < 256; ++r) s += x[(size_t)(r0 + r) * DIM + c];
    partial[blockIdx.y * DIM + c] = s;
}

// ---------------- Stage 2: finalize pooled mean ----------------
__global__ __launch_bounds__(256) void pooled_kernel(const float* __restrict__ partial,
                                                     float* __restrict__ pooled) {
    const int c = blockIdx.x * 256 + threadIdx.x;
    float s = 0.0f;
    for (int g = 0; g < 128; ++g) s += partial[g * DIM + c];
    pooled[c] = s * (1.0f / (float)SEQ);
}

// ---------------- Stage 3: complexity MLP -> dynamic k + scalar outputs ----------------
__global__ __launch_bounds__(128) void mlp_kernel(const float* __restrict__ pooled,
                                                  const float* __restrict__ w1,
                                                  const float* __restrict__ b1,
                                                  const float* __restrict__ w2,
                                                  const float* __restrict__ b2,
                                                  int* __restrict__ kscal,
                                                  float* __restrict__ out_tail) {
    __shared__ float red[128];
    const int t = threadIdx.x;                       // 128 hidden units
    float acc = b1[t];
    for (int i = 0; i < DIM; ++i) acc += pooled[i] * w1[i * 128 + t];
    float h = 0.5f * acc * (1.0f + erff(acc * 0.70710678118654752f));  // exact GELU
    red[t] = h * w2[t];
    __syncthreads();
    for (int off = 64; off > 0; off >>= 1) {
        if (t < off) red[t] += red[t + off];
        __syncthreads();
    }
    if (t == 0) {
        float z = red[0] + b2[0];
        float c = 1.0f / (1.0f + expf(-z));
        float ratio = 0.01f + 0.09f * c;
        int k = (int)(ratio * (float)DIM);           // truncation, like torch .int()
        k = k < 1 ? 1 : (k > KMAXC ? KMAXC : k);
        kscal[0] = k;
        out_tail[0] = c;
        out_tail[1] = ratio;
        out_tail[2] = (float)k;
    }
}

// ---------------- Stage 4: wg (K,N) f32 -> wgT (N,K) bf16, LDS tiled ----------------
__global__ __launch_bounds__(256) void wgt_kernel(const float* __restrict__ wg,
                                                  __bf16* __restrict__ wgT) {
    __shared__ float tile[32][33];
    const int tx = threadIdx.x & 31, ty = threadIdx.x >> 5;    // 32x8
    const int n0 = blockIdx.x * 32, k0 = blockIdx.y * 32;
    for (int i = 0; i < 4; ++i)
        tile[ty + 8 * i][tx] = wg[(size_t)(k0 + ty + 8 * i) * DIM + (n0 + tx)];
    __syncthreads();
    for (int i = 0; i < 4; ++i)
        wgT[(size_t)(n0 + ty + 8 * i) * DIM + (k0 + tx)] = (__bf16)tile[tx][ty + 8 * i];
}

// ---- Stage 5: 64-row WMMA GEMM (16 waves, 4Mx4N tiles/wave) + normalize + radix top-k ----
__global__ __launch_bounds__(512) void gate_kernel(const float* __restrict__ x,
                                                   const __bf16* __restrict__ wgT,
                                                   const float* __restrict__ bg,
                                                   const float* __restrict__ rmean,
                                                   const float* __restrict__ rvar,
                                                   const int* __restrict__ kptr,
                                                   float* __restrict__ out) {
    __shared__ float    imp[8 * DIM];      // 32 KB: 8 rows of importance at a time
    __shared__ unsigned histAll[8 * 256];  // 8 KB: one 256-bin histogram per select-wave
    __shared__ unsigned swctl[8 * 2];      // per-select-wave {prefix, kk} during passes
    __shared__ unsigned thrKeys[8];        // published per-row thresholds

    const int tid  = threadIdx.x;
    const int lane = tid & 31;
    const int wave = tid >> 5;                  // 16 waves; wave owns N-tiles 4w..4w+3
    const int m0   = blockIdx.x * 64;           // 64 rows per block (4 M-tiles)
    const int nlo  = lane & 15;
    const int hiA  = (lane >> 4) << 4;          // 16-bit A layout: K sub-offset
    const int hiB  = (lane >> 4) << 4;          // 16-bit B layout: K sub-offset

    const int kdyn = *kptr;

    v8f acc[4][4] = {};                          // [M-tile][N-tile], 128 VGPRs
    const float* xr[4];
#pragma unroll
    for (int g = 0; g < 4; ++g) xr[g] = x + (size_t)(m0 + g * 16 + nlo) * DIM;
    const __bf16* wgTw = wgT + (size_t)(wave * 4) * 16 * DIM + (size_t)nlo * DIM + hiB;

    for (int k0 = 0; k0 < DIM; k0 += 32) {
        // Four A fragments (16x32 bf16 each), fp32 -> bf16 on the fly.
        union AU { v16bf v; __bf16 e[16]; } au[4];
#pragma unroll
        for (int g = 0; g < 4; ++g) {
            const float4* p0 = (const float4*)(xr[g] + k0 + hiA);
            const float4 f0 = p0[0], f1 = p0[1];
            const float4* p1 = (const float4*)(xr[g] + k0 + 8 + hiA);
            const float4 f2 = p1[0], f3 = p1[1];
            au[g].e[0]  = (__bf16)f0.x; au[g].e[1]  = (__bf16)f0.y;
            au[g].e[2]  = (__bf16)f0.z; au[g].e[3]  = (__bf16)f0.w;
            au[g].e[4]  = (__bf16)f1.x; au[g].e[5]  = (__bf16)f1.y;
            au[g].e[6]  = (__bf16)f1.z; au[g].e[7]  = (__bf16)f1.w;
            au[g].e[8]  = (__bf16)f2.x; au[g].e[9]  = (__bf16)f2.y;
            au[g].e[10] = (__bf16)f2.z; au[g].e[11] = (__bf16)f2.w;
            au[g].e[12] = (__bf16)f3.x; au[g].e[13] = (__bf16)f3.y;
            au[g].e[14] = (__bf16)f3.z; au[g].e[15] = (__bf16)f3.w;
        }
        if (k0 + 32 < DIM) {
            __builtin_prefetch(xr[0] + k0 + 32 + hiA, 0, 0);
            __builtin_prefetch(xr[2] + k0 + 32 + hiA, 0, 0);
        }
        // Each B fragment feeds FOUR WMMAs (all M-tiles): 4x B reuse in registers.
#pragma unroll
        for (int t = 0; t < 4; ++t) {
            const v16bf bv = *(const v16bf*)(wgTw + (size_t)t * 16 * DIM + k0);
#pragma unroll
            for (int g = 0; g < 4; ++g)
                acc[g][t] = __builtin_amdgcn_wmma_f32_16x16x32_bf16(
                    false, au[g].v, false, bv, (short)0, acc[g][t], false, false);
        }
    }

    // Normalize importance IN PLACE: (|acc + bg| - mean) / (sqrt(var)+eps)
#pragma unroll
    for (int t = 0; t < 4; ++t) {
        const int n = (wave * 4 + t) * 16 + nlo;
        const float bgv = bg[n];
        const float mu  = rmean[n];
        const float inv = 1.0f / (sqrtf(rvar[n]) + 1e-6f);
#pragma unroll
        for (int g = 0; g < 4; ++g)
#pragma unroll
            for (int v = 0; v < 8; ++v)
                acc[g][t][v] = (fabsf(acc[g][t][v] + bgv) - mu) * inv;
    }

    // D layout: within an M-tile, lanes 0-15 hold rows 0-7 (VGPR v = row v),
    // lanes 16-31 hold rows 8-15. Eight phases of 8 rows; waves 0-7 each
    // radix-select one row (barrier-free), then all 16 waves apply the mask.
    unsigned* histw = &histAll[(wave & 7) * 256];

    for (int p = 0; p < 8; ++p) {               // phase p covers rows m0+p*8 .. +7
        const int g = p >> 1;                   // which M-tile
        if ((lane >> 4) == (p & 1)) {           // which half of the tile
#pragma unroll
            for (int t = 0; t < 4; ++t) {
                const int n = (wave * 4 + t) * 16 + nlo;
#pragma unroll
                for (int v = 0; v < 8; ++v) imp[v * DIM + n] = acc[g][t][v];
            }
        }
        __syncthreads();

        if (wave < 8) {
            const int r = wave;                 // this wave's row (0..7)
            const float* impr = &imp[r * DIM];
            // Exact kth-largest via 4-pass (MSB-first) radix select on sortable keys.
            unsigned prefix = 0;
            int kk = kdyn;
            for (int pass = 0; pass < 4; ++pass) {
                const int shift = 24 - 8 * pass;
#pragma unroll
                for (int i = 0; i < 8; ++i) histw[lane * 8 + i] = 0u;  // wave-local, in-order
                for (int i = 0; i < 32; ++i) {
                    const int j = i * 32 + lane;
                    const unsigned key = keyOf(impr[j]);
                    const bool match = (pass == 0) || ((key >> (shift + 8)) == prefix);
                    if (match) atomicAdd(&histw[(key >> shift) & 255u], 1u);
                }
                // descending suffix-sum over 256 bins: 8 bins per lane + wave scan
                unsigned c[8], csuf[8];
#pragma unroll
                for (int i = 0; i < 8; ++i) c[i] = histw[lane * 8 + i];
                unsigned tot = 0;
#pragma unroll
                for (int i = 7; i >= 0; --i) { tot += c[i]; csuf[i] = tot; }
                unsigned v = tot;               // inclusive suffix across lanes
                for (int off = 1; off < 32; off <<= 1) {
                    const unsigned u = __shfl_down(v, off, 32);
                    if (lane + off < 32) v += u;
                }
                const unsigned above = v - tot; // strictly-higher lanes' chunk sum
#pragma unroll
                for (int i = 0; i < 8; ++i) {   // unique crossing bin
                    const unsigned cum = above + csuf[i];
                    const unsigned nxt = cum - c[i];
                    if (cum >= (unsigned)kk && nxt < (unsigned)kk) {
                        swctl[r * 2 + 0] = (prefix << 8) | (unsigned)(lane * 8 + i);
                        swctl[r * 2 + 1] = (unsigned)kk - nxt;
                    }
                }
                prefix = swctl[r * 2 + 0];
                kk = (int)swctl[r * 2 + 1];
            }
            if (lane == 0) thrKeys[r] = prefix; // key of kth-largest for this row
        }
        __syncthreads();

        // apply mask: wave pair {w, w+8} splits row w&7 (512 elements each)
        {
            const int r = wave & 7;
            const unsigned thrKey = thrKeys[r];
            const float* impr = &imp[r * DIM];
            const int jbase = (wave >> 3) * 512;
            const int grow = m0 + p * 8 + r;
            const float* xg = x + (size_t)grow * DIM;
            float* og = out + (size_t)grow * DIM;
            for (int i = 0; i < 16; ++i) {
                const int j = jbase + i * 32 + lane;
                const float m = (keyOf(impr[j]) >= thrKey) ? 1.0f : 0.0f;
                og[j] = xg[j] * m;
            }
        }
        __syncthreads();                        // before imp is overwritten next phase
    }
}

// ---------------- launcher ----------------
extern "C" void kernel_launch(void* const* d_in, const int* in_sizes, int n_in,
                              void* d_out, int out_size, void* d_ws, size_t ws_size,
                              hipStream_t stream) {
    const float* x     = (const float*)d_in[0];
    const float* w1    = (const float*)d_in[1];
    const float* b1    = (const float*)d_in[2];
    const float* w2    = (const float*)d_in[3];
    const float* b2    = (const float*)d_in[4];
    const float* wg    = (const float*)d_in[5];
    const float* bg    = (const float*)d_in[6];
    const float* rmean = (const float*)d_in[7];
    const float* rvar  = (const float*)d_in[8];
    float* out = (float*)d_out;

    // workspace layout (~2.6 MB)
    char* ws = (char*)d_ws;
    __bf16* wgT     = (__bf16*)ws;                                  // 2 MB
    float*  partial = (float*)(ws + (2u << 20));                    // 512 KB
    float*  pooled  = (float*)(ws + (2u << 20) + (512u << 10));     // 4 KB
    int*    kscal   = (int*)  (ws + (2u << 20) + (512u << 10) + 4096);

    colsum_kernel<<<dim3(4, 128), 256, 0, stream>>>(x, partial);
    pooled_kernel<<<4, 256, 0, stream>>>(partial, pooled);
    mlp_kernel<<<1, 128, 0, stream>>>(pooled, w1, b1, w2, b2, kscal,
                                      out + (size_t)SEQ * DIM);
    wgt_kernel<<<dim3(32, 32), 256, 0, stream>>>(wg, wgT);
    gate_kernel<<<SEQ / 64, 512, 0, stream>>>(x, wgT, bg, rmean, rvar, kscal, out);
}
